// OLD_LSTM_Model_44049184587866
// MI455X (gfx1250) — matrix operand order repaired and checked by reading, not compiled, not run on hardware
//
#include <hip/hip_runtime.h>
#include <hip/hip_bf16.h>

// ---------------------------------------------------------------------------
// LSTMP stack (T=160, B=640, D=40, H=768, P=256, L=3) for gfx1250 (MI455X).
// bf16 WMMA (V_WMMA_F32_16X16X32_BF16) for both GEMMs, f32 state.
// Layers interleaved per timestep -> no [T,B,P] buffers.
// GEMM: per-block LDS staging of the 16-row A strip via async-to-LDS loads,
// register double-buffered K loop (loads for slice k+32 issued before the
// 4 WMMAs of slice k).
// ---------------------------------------------------------------------------

typedef __attribute__((ext_vector_type(8)))  __bf16 v8bf;
typedef __attribute__((ext_vector_type(16))) __bf16 v16bf;
typedef __attribute__((ext_vector_type(8)))  float  v8f;
typedef __attribute__((ext_vector_type(4)))  int    v4i_t;

#define BATCH 640
#define HID   768
#define PROJ  256
#define NGATE 3072          // 4*H
#define MAXKP 768           // max GEMM K (projection)
#define KPAD_EXTRA 8        // LDS row pad (bf16 elems) => +16B per row

#if defined(__has_builtin)
#if __has_builtin(__builtin_amdgcn_global_load_async_to_lds_b128)
#define HAVE_ASYNC_LDS 1
#endif
#if __has_builtin(__builtin_amdgcn_s_wait_asynccnt)
#define HAVE_WAIT_ASYNC_BUILTIN 1
#endif
#endif

#if defined(HAVE_ASYNC_LDS)
// pointer to int4 in global (device) AS, and in LDS AS
typedef __attribute__((address_space(1))) v4i_t* gv4i_p;
typedef __attribute__((address_space(3))) v4i_t* lv4i_p;
#endif

__device__ __forceinline__ void wait_asynccnt0() {
#if defined(HAVE_WAIT_ASYNC_BUILTIN)
    __builtin_amdgcn_s_wait_asynccnt(0);
#else
    asm volatile("s_wait_asynccnt 0x0" ::: "memory");
#endif
}

// ---------------------------------------------------------------------------
// Swizzle f32 weights [Kreal, N] row-major into bf16 WMMA B-fragment tiles.
// Tile = 32(K) x 16(N) bf16 = 512 elems; tile index = kt*ntilesN + nt.
// element (lane*16 + e) <-> k = kt*32 + (lane<16 ? e : 16+e),
//                           n = nt*16 + (lane & 15).   K >= Kreal -> 0.
// ---------------------------------------------------------------------------
__global__ void lstmp_swizzle_w(const float* __restrict__ W,
                                __bf16* __restrict__ out,
                                int Kreal, int Kpad, int N)
{
    int idx = blockIdx.x * blockDim.x + threadIdx.x;
    int total = Kpad * N;
    if (idx >= total) return;
    int tile   = idx >> 9;
    int within = idx & 511;
    int lane   = within >> 4;
    int e      = within & 15;
    int ntN    = N >> 4;
    int kt     = tile / ntN;
    int nt     = tile % ntN;
    int k = kt * 32 + ((lane < 16) ? e : (16 + e));
    int n = nt * 16 + (lane & 15);
    float v = (k < Kreal) ? W[(size_t)k * N + n] : 0.0f;
    out[idx] = (__bf16)v;
}

// ---------------------------------------------------------------------------
// Build A = [x_t | h_{t-1}] as bf16 [BATCH, Kpad] row-major, zero pad.
// ---------------------------------------------------------------------------
__global__ void lstmp_pack(const float* __restrict__ xsrc, int xStride, int din,
                           const float* __restrict__ hprev, int useH,
                           __bf16* __restrict__ Abf, int Kpad)
{
    int idx = blockIdx.x * blockDim.x + threadIdx.x;
    int total = BATCH * Kpad;
    if (idx >= total) return;
    int b = idx / Kpad;
    int k = idx % Kpad;
    float v = 0.0f;
    if (k < din)               v = xsrc[(size_t)b * xStride + k];
    else if (k < din + PROJ) { if (useH) v = hprev[b * PROJ + (k - din)]; }
    Abf[idx] = (__bf16)v;
}

// ---------------------------------------------------------------------------
// GEMM: out[640, N] = A[640, Kpad](bf16) @ Bsw(swizzled bf16) (+ bias).
// Block: (blockDim/32) waves sharing one 16-row M tile; A strip staged in LDS
// via async-to-LDS loads; each wave owns a 16x64 output tile (4 accumulators).
// K loop double-buffered: loads for slice k+32 precede WMMAs of slice k.
// Launch: grid = 40 * nChunks / wavesPerBlock.
// ---------------------------------------------------------------------------
__global__ __launch_bounds__(256) void lstmp_gemm_wmma(
    const __bf16* __restrict__ A,     // [640, Kpad] row-major
    const __bf16* __restrict__ Bsw,   // swizzled tiles
    const float*  __restrict__ bias,  // [N] or nullptr
    float* __restrict__ out,          // [640, N]
    int Kpad, int N, int nChunks)
{
    __shared__ __bf16 smemA[16 * (MAXKP + KPAD_EXTRA)];

    const int KpadP      = Kpad + KPAD_EXTRA;
    const int nchPerBlk  = blockDim.x >> 5;
    const int grpPerMt   = nChunks / nchPerBlk;
    const int mt         = blockIdx.x / grpPerMt;
    const int grp        = blockIdx.x % grpPerMt;
    const int waveInBlk  = threadIdx.x >> 5;
    const int lane       = threadIdx.x & 31;
    const int nc         = grp * nchPerBlk + waveInBlk;
    const int m0         = mt << 4;
    const int n0         = nc << 6;

    // ---- stage A strip (rows m0..m0+15, Kpad elems each) into LDS ----
    {
        const int rowChunks = Kpad >> 3;              // 16B chunks per row
        const int total     = rowChunks << 4;         // 16 rows
        for (int c = threadIdx.x; c < total; c += blockDim.x) {
            int row = c / rowChunks;
            int kc  = (c % rowChunks) << 3;
            const __bf16* src = A + (size_t)(m0 + row) * Kpad + kc;
            __bf16*       dst = smemA + row * KpadP + kc;
#if defined(HAVE_ASYNC_LDS)
            __builtin_amdgcn_global_load_async_to_lds_b128(
                (gv4i_p)src, (lv4i_p)dst, 0, 0);
#else
            *(v8bf*)dst = *(const v8bf*)src;
#endif
        }
#if defined(HAVE_ASYNC_LDS)
        wait_asynccnt0();
#endif
        __syncthreads();
    }

    // ---- per-wave fragment pointers ----
    const int hiHalf = lane >> 4;                     // 0: lanes 0-15, 1: 16-31
    const int koff   = hiHalf << 4;                   // A K sub-offset (0 or 16)
    const __bf16* aptr = smemA + (lane & 15) * KpadP + koff;

    const int ntilesN   = N >> 4;
    const size_t ktStride = (size_t)ntilesN << 9;     // elems per K-slice
    const __bf16* bbase = Bsw + ((size_t)(n0 >> 4) << 9) + (lane << 4);

    v8f acc[4] = {v8f{}, v8f{}, v8f{}, v8f{}};

#define LOAD_A(kb, lo, hi)                                                     \
    do {                                                                       \
        lo = *(const v8bf*)(aptr + (kb));                                      \
        hi = *(const v8bf*)(aptr + (kb) + 8);                                  \
    } while (0)

#define LOAD_B(kb, B0, B1, B2, B3)                                             \
    do {                                                                       \
        const __bf16* bt = bbase + (size_t)((kb) >> 5) * ktStride;             \
        B0 = *(const v16bf*)(bt);                                              \
        B1 = *(const v16bf*)(bt + 512);                                        \
        B2 = *(const v16bf*)(bt + 1024);                                       \
        B3 = *(const v16bf*)(bt + 1536);                                       \
    } while (0)

#define WMMA4(lo, hi, B0, B1, B2, B3)                                          \
    do {                                                                       \
        v16bf a_;                                                              \
        _Pragma("unroll")                                                      \
        for (int i_ = 0; i_ < 8; ++i_) { a_[i_] = lo[i_]; a_[8 + i_] = hi[i_]; } \
        acc[0] = __builtin_amdgcn_wmma_f32_16x16x32_bf16(false, a_, false, B0, (short)0, acc[0], false, false); \
        acc[1] = __builtin_amdgcn_wmma_f32_16x16x32_bf16(false, a_, false, B1, (short)0, acc[1], false, false); \
        acc[2] = __builtin_amdgcn_wmma_f32_16x16x32_bf16(false, a_, false, B2, (short)0, acc[2], false, false); \
        acc[3] = __builtin_amdgcn_wmma_f32_16x16x32_bf16(false, a_, false, B3, (short)0, acc[3], false, false); \
    } while (0)

    v8bf  aLo0, aHi0, aLo1, aHi1;
    v16bf b00, b01, b02, b03, b10, b11, b12, b13;

    LOAD_A(0, aLo0, aHi0);
    LOAD_B(0, b00, b01, b02, b03);

    // Kpad/32 is even for all shapes used (10, 16, 24) -> unroll by 2.
    for (int kb = 0; kb + 64 <= Kpad; kb += 64) {
        LOAD_A(kb + 32, aLo1, aHi1);
        LOAD_B(kb + 32, b10, b11, b12, b13);
        if (kb + 96 < Kpad)
            __builtin_prefetch(bbase + (size_t)((kb + 96) >> 5) * ktStride, 0, 0);
        WMMA4(aLo0, aHi0, b00, b01, b02, b03);

        if (kb + 64 < Kpad) {
            LOAD_A(kb + 64, aLo0, aHi0);
            LOAD_B(kb + 64, b00, b01, b02, b03);
        }
        WMMA4(aLo1, aHi1, b10, b11, b12, b13);
    }

#undef LOAD_A
#undef LOAD_B
#undef WMMA4

    // ---- epilogue: lane&15 = column, VGPR r = row m0 + r + 8*hiHalf ----
    const int colBase = n0 + (lane & 15);
    const int rbase   = m0 + (hiHalf << 3);
    #pragma unroll
    for (int j = 0; j < 4; ++j) {
        int col  = colBase + (j << 4);
        float bv = bias ? bias[col] : 0.0f;
        #pragma unroll
        for (int r = 0; r < 8; ++r)
            out[(size_t)(rbase + r) * N + col] = acc[j][r] + bv;
    }
}

// ---------------------------------------------------------------------------
// Cell update: c' = sig(f+1)*c + sig(i)*tanh(j); g = sig(o)*tanh(c') (bf16).
// z gate order (i, j, f, o).
// ---------------------------------------------------------------------------
__global__ void lstmp_cell(const float* __restrict__ z,
                           float* __restrict__ c,
                           __bf16* __restrict__ gbf)
{
    int idx = blockIdx.x * blockDim.x + threadIdx.x;
    if (idx >= BATCH * HID) return;
    int b  = idx / HID;
    int hh = idx % HID;
    const float* zb = z + (size_t)b * NGATE;
    float gi = zb[hh];
    float gj = zb[HID + hh];
    float gf = zb[2 * HID + hh];
    float go = zb[3 * HID + hh];
    float cv = c[idx];
    float sf = 1.0f / (1.0f + __expf(-(gf + 1.0f)));
    float si = 1.0f / (1.0f + __expf(-gi));
    float so = 1.0f / (1.0f + __expf(-go));
    cv = sf * cv + si * tanhf(gj);
    c[idx]   = cv;
    gbf[idx] = (__bf16)(so * tanhf(cv));
}

// ---------------------------------------------------------------------------
// out[b,:] = h[b,:] / (||h[b,:]|| + 1e-6).
// ---------------------------------------------------------------------------
__global__ void lstmp_normalize(const float* __restrict__ h, float* __restrict__ out)
{
    __shared__ float red[256];
    int b = blockIdx.x;
    int t = threadIdx.x;
    float v = h[(size_t)b * PROJ + t];
    red[t] = v * v;
    __syncthreads();
    for (int s = 128; s > 0; s >>= 1) {
        if (t < s) red[t] += red[t + s];
        __syncthreads();
    }
    float norm = sqrtf(red[0]) + 1e-6f;
    out[(size_t)b * PROJ + t] = v / norm;
}

// ---------------------------------------------------------------------------
// Host orchestration. d_in: x, W0, b0, Wp0, W1, b1, Wp1, W2, b2, Wp2 (f32).
// ---------------------------------------------------------------------------
extern "C" void kernel_launch(void* const* d_in, const int* in_sizes, int n_in,
                              void* d_out, int out_size, void* d_ws, size_t ws_size,
                              hipStream_t stream)
{
    (void)in_sizes; (void)n_in; (void)out_size; (void)ws_size;

    const float* x = (const float*)d_in[0];
    const float* W [3] = { (const float*)d_in[1], (const float*)d_in[4], (const float*)d_in[7] };
    const float* bb[3] = { (const float*)d_in[2], (const float*)d_in[5], (const float*)d_in[8] };
    const float* Wp[3] = { (const float*)d_in[3], (const float*)d_in[6], (const float*)d_in[9] };

    const int T = 160, D = 40;
    const int din  [3] = { D, PROJ, PROJ };
    const int Kreal[3] = { D + PROJ, 2 * PROJ, 2 * PROJ };   // 296, 512, 512
    const int Kpad [3] = { 320, 512, 512 };

    // ---- carve workspace (256B aligned) ----
    char*  ws  = (char*)d_ws;
    size_t off = 0;
    auto carve = [&](size_t bytes) -> char* {
        off = (off + 255) & ~(size_t)255;
        char* p = ws + off;
        off += bytes;
        return p;
    };
    __bf16* Wsw [3]; __bf16* Wpsw[3];
    for (int l = 0; l < 3; ++l) Wsw [l] = (__bf16*)carve((size_t)Kpad[l] * NGATE * sizeof(__bf16));
    for (int l = 0; l < 3; ++l) Wpsw[l] = (__bf16*)carve((size_t)HID * PROJ * sizeof(__bf16));
    __bf16* Abf = (__bf16*)carve((size_t)BATCH * 512  * sizeof(__bf16));
    float*  z   = (float* )carve((size_t)BATCH * NGATE * sizeof(float));
    float*  c0  = (float* )carve((size_t)3 * BATCH * HID * sizeof(float));
    __bf16* gbf = (__bf16*)carve((size_t)BATCH * HID * sizeof(__bf16));
    float*  hf [3];
    for (int l = 0; l < 3; ++l) hf[l] = (float*)carve((size_t)BATCH * PROJ * sizeof(float));
    float* cs[3] = { c0, c0 + (size_t)BATCH * HID, c0 + (size_t)2 * BATCH * HID };

    // ---- one-time weight swizzle ----
    for (int l = 0; l < 3; ++l) {
        int totW = Kpad[l] * NGATE;
        lstmp_swizzle_w<<<(totW + 255) / 256, 256, 0, stream>>>(W[l], Wsw[l], Kreal[l], Kpad[l], NGATE);
        int totP = HID * PROJ;
        lstmp_swizzle_w<<<(totP + 255) / 256, 256, 0, stream>>>(Wp[l], Wpsw[l], HID, HID, PROJ);
    }

    (void)hipMemsetAsync(c0, 0, (size_t)3 * BATCH * HID * sizeof(float), stream);

    // gate:  40 M-tiles * 48 chunks, 8 waves/block -> 240 blocks of 256
    // proj:  40 M-tiles *  4 chunks, 4 waves/block ->  40 blocks of 128
    const int gateBlocks = (BATCH / 16) * (NGATE / 64) / 8;
    const int projBlocks = (BATCH / 16) * (PROJ  / 64) / 4;

    for (int t = 0; t < T; ++t) {
        for (int l = 0; l < 3; ++l) {
            const float* xsrc = (l == 0) ? (x + (size_t)t * BATCH * D) : hf[l - 1];
            int xStride = (l == 0) ? D : PROJ;

            int packTot = BATCH * Kpad[l];
            lstmp_pack<<<(packTot + 255) / 256, 256, 0, stream>>>(
                xsrc, xStride, din[l], hf[l], (t > 0) ? 1 : 0, Abf, Kpad[l]);

            lstmp_gemm_wmma<<<gateBlocks, 256, 0, stream>>>(
                Abf, Wsw[l], bb[l], z, Kpad[l], NGATE, NGATE / 64);

            lstmp_cell<<<(BATCH * HID + 255) / 256, 256, 0, stream>>>(z, cs[l], gbf);

            lstmp_gemm_wmma<<<projBlocks, 128, 0, stream>>>(
                gbf, Wpsw[l], nullptr, hf[l], HID, PROJ, PROJ / 64);
        }
    }

    lstmp_normalize<<<BATCH, 256, 0, stream>>>(hf[2], (float*)d_out);
}
